// MultiHeadAttention_68169720922469
// MI455X (gfx1250) — matrix operand order repaired and checked by reading, not compiled, not run on hardware
//
#include <hip/hip_runtime.h>

// ---------------------------------------------------------------------------
// MI455X (gfx1250, wave32) multi-head attention:
//   QKV projections (WMMA bf16) -> causal flash attention, transposed-score
//   formulation (WMMA bf16)     -> output projection (WMMA bf16, f32 out)
// ---------------------------------------------------------------------------

typedef __attribute__((ext_vector_type(16))) __bf16 v16bf;
typedef __attribute__((ext_vector_type(8)))  float  v8f;

struct U4 { unsigned int x, y, z, w; };            // 16-byte POD chunk
struct U2 { unsigned int x, y; };                  // 8-byte POD chunk

union BF16x16 {                                    // one WMMA bf16 operand
    v16bf          v;
    U4             q[2];
    unsigned short s[16];
};

__device__ __forceinline__ unsigned short f2bf(float f) {
    __bf16 h = (__bf16)f;                          // hardware v_cvt, RNE
    return __builtin_bit_cast(unsigned short, h);
}
__device__ __forceinline__ unsigned int pk2bf(float a, float b) {
    return (unsigned int)f2bf(a) | ((unsigned int)f2bf(b) << 16);
}

#define SEQ    2048
#define DMODEL 1024
#define HEADS  16
#define HD     64

// ---------------------------------------------------------------------------
// Tiled GEMM: C[M,N] = (A[M,K] @ W[K,N] + bias) * scale
//   A: f32 or bf16(ushort) row-major; W: f32 row-major (bf16-converted)
//   SCATTER: write bf16 into [b][h][s][64]; else write f32 row-major
// 256 thr = 8 waves; tile 128x128, BK=32; each wave owns a 16x128 slab.
// ---------------------------------------------------------------------------
template <bool A_F32, bool SCATTER>
__global__ __launch_bounds__(256) void gemm_wmma(const void* __restrict__ Ap,
                                                 const float* __restrict__ W,
                                                 const float* __restrict__ bias,
                                                 void* __restrict__ Out,
                                                 int M, int K, int N,
                                                 float scale) {
    __shared__ alignas(16) unsigned short As[128 * 32];  // A tile, bf16
    __shared__ alignas(16) unsigned short Bt[128 * 40];  // W tile, transposed+pad

    const int tid  = threadIdx.x;
    const int wave = tid >> 5;
    const int lane = tid & 31;
    const int hi   = lane >> 4;
    const int lo   = lane & 15;
    const int m0   = blockIdx.y * 128;
    const int n0   = blockIdx.x * 128;

    v8f acc[8];
#pragma unroll
    for (int t = 0; t < 8; ++t) acc[t] = {};

    for (int kb = 0; kb < K; kb += 32) {
        // ---- stage A tile (128x32): 2 threads per row, 16 elems each ----
        {
            const int row = tid >> 1;
            const int col = (tid & 1) * 16;
            BF16x16 pk;
            if constexpr (A_F32) {
                const float* src = (const float*)Ap + (size_t)(m0 + row) * K + kb + col;
#pragma unroll
                for (int i = 0; i < 16; ++i) pk.s[i] = f2bf(src[i]);
            } else {
                const unsigned short* src =
                    (const unsigned short*)Ap + (size_t)(m0 + row) * K + kb + col;
                pk.q[0] = *(const U4*)(src);
                pk.q[1] = *(const U4*)(src + 8);
            }
            *(U4*)&As[row * 32 + col]     = pk.q[0];
            *(U4*)&As[row * 32 + col + 8] = pk.q[1];
        }
        // ---- stage W tile transposed: Bt[n][k], stride 40, b64 stores ----
        {
            const int k0 = (tid & 7) * 4;          // k quad: 0,4,..,28
            const int nq = (tid >> 3) * 4;         // n quad: 0,4,..,124
            float v[4][4];
#pragma unroll
            for (int j = 0; j < 4; ++j) {
                const float* src = W + (size_t)(kb + k0 + j) * N + n0 + nq;
#pragma unroll
                for (int i = 0; i < 4; ++i) v[j][i] = src[i];
            }
#pragma unroll
            for (int i = 0; i < 4; ++i) {
                U2 u;
                u.x = pk2bf(v[0][i], v[1][i]);
                u.y = pk2bf(v[2][i], v[3][i]);
                *(U2*)&Bt[(nq + i) * 40 + k0] = u;   // 4 ushorts: k0..k0+3
            }
        }
        __syncthreads();

        // ---- compute: A frag (16x32) x 8 B frags (32x16) per wave ----
        const int arow = wave * 16 + lo;
        BF16x16 af;
        af.q[0] = *(const U4*)&As[arow * 32 + hi * 8];        // K = 8*hi..+7
        af.q[1] = *(const U4*)&As[arow * 32 + 16 + hi * 8];   // K = 16+8*hi..+7
#pragma unroll
        for (int t = 0; t < 8; ++t) {
            BF16x16 bf_;
            const int n = t * 16 + lo;
            bf_.q[0] = *(const U4*)&Bt[n * 40 + hi * 8];      // K = 8*hi..+7
            bf_.q[1] = *(const U4*)&Bt[n * 40 + 16 + hi * 8]; // K = 16+8*hi..+7
            acc[t] = __builtin_amdgcn_wmma_f32_16x16x32_bf16(
                false, af.v, false, bf_.v, (short)0, acc[t], false, false);
        }
        __syncthreads();
    }

    // ---- epilogue: bias, scale, store ----
    const int mbase = m0 + wave * 16 + 8 * hi;     // C/D row = r + 8*hi
#pragma unroll
    for (int t = 0; t < 8; ++t) {
        const int n  = n0 + t * 16 + lo;
        const float bv = bias[n];
#pragma unroll
        for (int r = 0; r < 8; ++r) {
            const int   m = mbase + r;
            const float v = (acc[t][r] + bv) * scale;
            if constexpr (SCATTER) {
                const int h = n >> 6, d = n & 63;
                const int b = m >> 11, s = m & 2047;
                ((unsigned short*)Out)[(((size_t)b * HEADS + h) * SEQ + s) * HD + d] =
                    f2bf(v);
            } else {
                ((float*)Out)[(size_t)m * N + n] = v;
            }
        }
    }
}

// ---------------------------------------------------------------------------
// Causal flash attention, transposed-score formulation.
//   S^T = K Q^T  (K is the WMMA A operand, Q^T the B operand)
//   O^T = V^T P^T
// All 32 key-scores of one query live in one lane (split across the two
// 16-lane halves) -> row max/sum = in-lane tree + ONE shfl_xor(16).
// One (b,h) per blockIdx.y; 64 query rows per block (4 waves x 16).
// Q is pre-scaled by 1/sqrt(HD) in the projection epilogue.
// ---------------------------------------------------------------------------
__global__ __launch_bounds__(128) void flash_attn(const unsigned short* __restrict__ Qg,
                                                  const unsigned short* __restrict__ Kg,
                                                  const unsigned short* __restrict__ Vg,
                                                  unsigned short* __restrict__ ctx) {
    __shared__ alignas(16) unsigned short Ks[32 * 64];      // K tile [key][dim]
    __shared__ alignas(16) unsigned short Vt[64 * 40];      // V^T tile [dim][key]+pad
    __shared__ alignas(16) unsigned short Ps[4][16 * 32];   // per-wave P^T [query][key]

    const int tid  = threadIdx.x;
    const int wave = tid >> 5;
    const int lane = tid & 31;
    const int hi   = lane >> 4;
    const int lo   = lane & 15;
    const int bh   = blockIdx.y;                  // b*HEADS + h
    const int q0   = blockIdx.x * 64;
    const int qbase = q0 + wave * 16;
    const int qg   = qbase + lo;                  // this lane's query row

    // ---- Q^T fragments (B operand): b[e] = Q[query=lo][c*32 + 16*hi + e] ----
    BF16x16 qb[2];
    {
        const unsigned short* qrow = Qg + ((size_t)bh * SEQ + qg) * HD;
#pragma unroll
        for (int c = 0; c < 2; ++c) {
            qb[c].q[0] = *(const U4*)(qrow + c * 32 + hi * 16);
            qb[c].q[1] = *(const U4*)(qrow + c * 32 + hi * 16 + 8);
        }
    }

    float mi = -__builtin_inff(), li = 0.0f;      // per-query (per-lane) stats
    v8f   o[4];                                   // O^T: dim = t*16 + r + 8*hi
#pragma unroll
    for (int t = 0; t < 4; ++t) o[t] = {};

    const int nkb = (q0 + 64) >> 5;               // causal: 32-key blocks

    for (int kb = 0; kb < nkb; ++kb) {
        const int kk0 = kb * 32;
        // ---- cooperative stage: K row-major, V transposed (packed b32) ----
        {
            const int row = tid >> 2;             // key 0..31
            const int seg = (tid & 3) * 16;
            const unsigned short* ksrc = Kg + ((size_t)bh * SEQ + kk0 + row) * HD + seg;
            *(U4*)&Ks[row * 64 + seg]     = *(const U4*)(ksrc);
            *(U4*)&Ks[row * 64 + seg + 8] = *(const U4*)(ksrc + 8);

            const int kp = (tid & 15) * 2;        // key pair
            const int dsg = (tid >> 4) * 8;       // dim segment 0..56
            const unsigned short* v0 = Vg + ((size_t)bh * SEQ + kk0 + kp) * HD + dsg;
            BF16x16 pv;
            pv.q[0] = *(const U4*)(v0);           // key kp,   dims dsg..+7
            pv.q[1] = *(const U4*)(v0 + HD);      // key kp+1, dims dsg..+7
#pragma unroll
            for (int i = 0; i < 8; ++i) {
                unsigned int u = (unsigned int)pv.s[i] | ((unsigned int)pv.s[8 + i] << 16);
                *(unsigned int*)&Vt[(dsg + i) * 40 + kp] = u;
            }
        }
        __syncthreads();

        // ---- S^T = K Q^T : two 16-key tiles x two 32-dim chunks ----
        v8f sa[2];
        sa[0] = {}; sa[1] = {};
#pragma unroll
        for (int c = 0; c < 2; ++c) {
#pragma unroll
            for (int t = 0; t < 2; ++t) {
                BF16x16 ka;                       // A: m = key = t*16+lo
                const int krow = (t * 16 + lo) * 64 + c * 32;
                ka.q[0] = *(const U4*)&Ks[krow + hi * 16];
                ka.q[1] = *(const U4*)&Ks[krow + hi * 16 + 8];
                sa[t] = __builtin_amdgcn_wmma_f32_16x16x32_bf16(
                    false, ka.v, false, qb[c].v, (short)0, sa[t], false, false);
            }
        }

        // ---- causal mask + in-lane softmax (one shuffle per reduction) ----
        float p[2][8];
        float rm = -__builtin_inff();
#pragma unroll
        for (int t = 0; t < 2; ++t)
#pragma unroll
            for (int r = 0; r < 8; ++r) {
                const int key = kk0 + t * 16 + r + 8 * hi;
                const float v = (key <= qg) ? sa[t][r] : -__builtin_inff();
                p[t][r] = v;
                rm = fmaxf(rm, v);
            }
        rm = fmaxf(rm, __shfl_xor(rm, 16, 32));           // join the two halves
        const float mnew = fmaxf(mi, rm);
        float rs = 0.0f;
#pragma unroll
        for (int t = 0; t < 2; ++t)
#pragma unroll
            for (int r = 0; r < 8; ++r) {
                p[t][r] = __expf(p[t][r] - mnew);
                rs += p[t][r];
            }
        rs += __shfl_xor(rs, 16, 32);
        const float alpha = __expf(mi - mnew);
        li = li * alpha + rs;
        mi = mnew;
#pragma unroll
        for (int t = 0; t < 4; ++t) o[t] *= alpha;

        // ---- stage P^T [query][key], packed b128: keys t*16+8*hi..+7 ----
#pragma unroll
        for (int t = 0; t < 2; ++t) {
            U4 u;
            u.x = pk2bf(p[t][0], p[t][1]);
            u.y = pk2bf(p[t][2], p[t][3]);
            u.z = pk2bf(p[t][4], p[t][5]);
            u.w = pk2bf(p[t][6], p[t][7]);
            *(U4*)&Ps[wave][lo * 32 + t * 16 + hi * 8] = u;
        }
        asm volatile("s_wait_dscnt 0" ::: "memory");      // wave-private transpose

        BF16x16 pb;                                       // B: K' = key = 16*hi+e
        pb.q[0] = *(const U4*)&Ps[wave][lo * 32 + hi * 16];
        pb.q[1] = *(const U4*)&Ps[wave][lo * 32 + hi * 16 + 8];

        // ---- O^T += V^T P^T : 4 dim tiles ----
#pragma unroll
        for (int t = 0; t < 4; ++t) {
            BF16x16 va;                                   // A: m = dim = t*16+lo
            const int vrow = (t * 16 + lo) * 40;
            va.q[0] = *(const U4*)&Vt[vrow + hi * 16];
            va.q[1] = *(const U4*)&Vt[vrow + hi * 16 + 8];
            o[t] = __builtin_amdgcn_wmma_f32_16x16x32_bf16(
                false, va.v, false, pb.v, (short)0, o[t], false, false);
        }
        __syncthreads();
    }

    // ---- normalize, pack, write ctx [b][s][h][d] with b128 stores ----
    const int b = bh >> 4, h = bh & 15;
    const float inv = 1.0f / li;
    unsigned short* crow = ctx + (((size_t)b * SEQ + qg) * HEADS + h) * HD;
#pragma unroll
    for (int t = 0; t < 4; ++t) {
        U4 u;
        u.x = pk2bf(o[t][0] * inv, o[t][1] * inv);
        u.y = pk2bf(o[t][2] * inv, o[t][3] * inv);
        u.z = pk2bf(o[t][4] * inv, o[t][5] * inv);
        u.w = pk2bf(o[t][6] * inv, o[t][7] * inv);
        *(U4*)(crow + t * 16 + hi * 8) = u;               // dims t*16+8*hi..+7
    }
}

// ---------------------------------------------------------------------------
extern "C" void kernel_launch(void* const* d_in, const int* in_sizes, int n_in,
                              void* d_out, int out_size, void* d_ws, size_t ws_size,
                              hipStream_t stream) {
    const float* Xq = (const float*)d_in[0];
    const float* Xk = (const float*)d_in[1];
    const float* Xv = (const float*)d_in[2];
    const float* Wq = (const float*)d_in[3];
    const float* bq = (const float*)d_in[4];
    const float* Wk = (const float*)d_in[5];
    const float* bk = (const float*)d_in[6];
    const float* Wv = (const float*)d_in[7];
    const float* bv = (const float*)d_in[8];
    const float* Wo = (const float*)d_in[9];
    const float* bo = (const float*)d_in[10];

    const size_t NQKV = (size_t)4 * HEADS * SEQ * HD;
    unsigned short* Qw = (unsigned short*)d_ws;
    unsigned short* Kw = Qw + NQKV;
    unsigned short* Vw = Kw + NQKV;
    unsigned short* Cw = Vw + NQKV;                     // ctx bf16 [B,S,D]

    const int M = 4 * SEQ;                              // 8192
    dim3 gg(DMODEL / 128, M / 128);                     // (8, 64)

    const float qscale = 0.125f;                        // 1/sqrt(HD)
    gemm_wmma<true, true><<<gg, 256, 0, stream>>>(Xq, Wq, bq, Qw, M, DMODEL, DMODEL, qscale);
    gemm_wmma<true, true><<<gg, 256, 0, stream>>>(Xk, Wk, bk, Kw, M, DMODEL, DMODEL, 1.0f);
    gemm_wmma<true, true><<<gg, 256, 0, stream>>>(Xv, Wv, bv, Vw, M, DMODEL, DMODEL, 1.0f);

    flash_attn<<<dim3(SEQ / 64, 4 * HEADS), 128, 0, stream>>>(Qw, Kw, Vw, Cw);

    gemm_wmma<false, false><<<gg, 256, 0, stream>>>(Cw, Wo, bo, d_out, M, DMODEL, DMODEL, 1.0f);
}
